// LearnableThresholdDeltaReg_35588099014898
// MI455X (gfx1250) — matrix operand order repaired and checked by reading, not compiled, not run on hardware
//
#include <hip/hip_runtime.h>
#include <hip/hip_bf16.h>
#include <hip/hip_fp16.h>

typedef __attribute__((ext_vector_type(16))) _Float16 v16h;
typedef __attribute__((ext_vector_type(8)))  float    v8f;

#define H_  64
#define V_  64
#define FF_ 128
#define B_  128
#define L_  1024
#define T_  (L_ - 1)

// ---------------- WMMA fragment loaders (CDNA5 16x16x32 f16, wave32) -----
// A-matrix 16x32 f16 (ISA 7.12.2): lane l -> m = l&15; hi = l>>4;
// vgpr pair v: k = 16*(v>>2) + 8*hi + 2*(v&3) + {0,1}
__device__ __forceinline__ v16h frag_a_f32(const float* src, int ld, int kb, int lane) {
  int m = lane & 15, hi = lane >> 4;
  v16h a;
#pragma unroll
  for (int vv = 0; vv < 8; ++vv) {
    int k0 = ((vv >> 2) << 4) + (hi << 3) + ((vv & 3) << 1);
    a[2 * vv]     = (_Float16)src[m * ld + kb + k0];
    a[2 * vv + 1] = (_Float16)src[m * ld + kb + k0 + 1];
  }
  return a;
}
__device__ __forceinline__ v16h frag_a_f16(const _Float16* src, int ld, int kb, int lane) {
  int m = lane & 15, hi = lane >> 4;
  v16h a;
#pragma unroll
  for (int vv = 0; vv < 8; ++vv) {
    int k0 = ((vv >> 2) << 4) + (hi << 3) + ((vv & 3) << 1);
    a[2 * vv]     = src[m * ld + kb + k0];
    a[2 * vv + 1] = src[m * ld + kb + k0 + 1];
  }
  return a;
}
// B-matrix 32x16 f16: lane l -> n = l&15; k = 16*(v>>2) + 8*(l>>4) + 2*(v&3) + h.
// Element (k,n) of X @ W.T with row-major W(out,in) is W[n][k].
__device__ __forceinline__ v16h frag_b_w(const _Float16* Wh, int ld, int nb, int kb, int lane) {
  int n = lane & 15, hi = lane >> 4;
  v16h b;
#pragma unroll
  for (int vv = 0; vv < 8; ++vv) {
    int k0 = ((vv >> 2) << 4) + (hi << 3) + ((vv & 3) << 1);
    b[2 * vv]     = Wh[(nb + n) * ld + kb + k0];
    b[2 * vv + 1] = Wh[(nb + n) * ld + kb + k0 + 1];
  }
  return b;
}

// ---------------- Kernel W: convert weights to f16 ------------------------
__global__ void wconv_kernel(const float* __restrict__ W1, const float* __restrict__ W2,
                             const float* __restrict__ Wk, const float* __restrict__ Wv,
                             _Float16* __restrict__ W1h, _Float16* __restrict__ W2h,
                             _Float16* __restrict__ Wkh, _Float16* __restrict__ Wvh) {
  int idx = blockIdx.x * blockDim.x + threadIdx.x;
  if (idx < 8192)        W1h[idx]         = (_Float16)W1[idx];
  else if (idx < 16384)  W2h[idx - 8192]  = (_Float16)W2[idx - 8192];
  else if (idx < 20480)  Wkh[idx - 16384] = (_Float16)Wk[idx - 16384];
  else if (idx < 24576)  Wvh[idx - 20480] = (_Float16)Wv[idx - 20480];
}

// ---------------- Kernel A: embed + MLP + LN + k/v proj (WMMA) ------------
// One wave (32 threads) per 16-token tile. 48 v_wmma per tile.
__global__ __launch_bounds__(32) void fused_token_kernel(
    const int* __restrict__ seq, const float* __restrict__ embed,
    const float* __restrict__ b1, const float* __restrict__ b2,
    const float* __restrict__ ln_g, const float* __restrict__ ln_b,
    const _Float16* __restrict__ W1h, const _Float16* __restrict__ W2h,
    const _Float16* __restrict__ Wkh, const _Float16* __restrict__ Wvh,
    float* __restrict__ kn_ws, float* __restrict__ v_ws,
    float* __restrict__ vnorm_ws, float* __restrict__ hlast) {
  __shared__ float    ldsH[16 * 64];     // embed h, then x, then k
  __shared__ _Float16 ldsA1[16 * 128];   // relu MLP hidden (f16)
  __shared__ float    ldsHln[16 * 64];   // post-LN h
  __shared__ float    ldsV[16 * 64];     // v
  __shared__ float    ldsS[16];          // 1/||k||
  __shared__ int      ldsTok[16];

  const int lane = threadIdx.x;
  const int blk  = blockIdx.x;
  const int b    = blk >> 6;             // 64 tiles of 16 tokens per batch
  const int p0   = (blk & 63) << 4;
  const int n16  = lane & 15, hi = lane >> 4;

  if (lane < 16) ldsTok[lane] = seq[b * L_ + p0 + lane];
  __syncthreads();
  for (int idx = lane; idx < 16 * 64; idx += 32) {
    int r = idx >> 6, c = idx & 63;
    ldsH[idx] = embed[ldsTok[r] * H_ + c];
  }
  __syncthreads();

  // GEMM1: (16x64) @ W1.T(64x128) -> relu -> ldsA1 (f16)
  v16h A0 = frag_a_f32(ldsH, 64, 0, lane);
  v16h A1 = frag_a_f32(ldsH, 64, 32, lane);
#pragma unroll
  for (int j = 0; j < 8; ++j) {
    v8f c = {};
    v16h Bf0 = frag_b_w(W1h, 64, j * 16, 0, lane);
    c = __builtin_amdgcn_wmma_f32_16x16x32_f16(false, A0, false, Bf0, (short)0, c, false, false);
    v16h Bf1 = frag_b_w(W1h, 64, j * 16, 32, lane);
    c = __builtin_amdgcn_wmma_f32_16x16x32_f16(false, A1, false, Bf1, (short)0, c, false, false);
    float bias = b1[j * 16 + n16];
#pragma unroll
    for (int r = 0; r < 8; ++r) {
      int m = r + (hi << 3);
      float val = c[r] + bias;
      val = val > 0.f ? val : 0.f;
      ldsA1[m * 128 + j * 16 + n16] = (_Float16)val;
    }
  }
  __syncthreads();

  // GEMM2: (16x128) @ W2.T(128x64), residual x = h + ff + b2 into ldsH
  v16h A2[4];
#pragma unroll
  for (int t = 0; t < 4; ++t) A2[t] = frag_a_f16(ldsA1, 128, 32 * t, lane);
#pragma unroll
  for (int j = 0; j < 4; ++j) {
    v8f c = {};
#pragma unroll
    for (int t = 0; t < 4; ++t) {
      v16h Bf = frag_b_w(W2h, 128, j * 16, 32 * t, lane);
      c = __builtin_amdgcn_wmma_f32_16x16x32_f16(false, A2[t], false, Bf, (short)0, c, false, false);
    }
    float bias = b2[j * 16 + n16];
#pragma unroll
    for (int r = 0; r < 8; ++r) {
      int m = r + (hi << 3);
      int col = j * 16 + n16;
      ldsH[m * 64 + col] = ldsH[m * 64 + col] + c[r] + bias;
    }
  }
  __syncthreads();

  // LayerNorm: one lane per row
  if (lane < 16) {
    float s = 0.f, s2 = 0.f;
#pragma unroll 8
    for (int c = 0; c < 64; ++c) { float x = ldsH[lane * 64 + c]; s += x; s2 += x * x; }
    float mu   = s * (1.f / 64.f);
    float var  = s2 * (1.f / 64.f) - mu * mu;
    float rstd = rsqrtf(var + 1e-5f);
#pragma unroll 8
    for (int c = 0; c < 64; ++c) {
      float x = ldsH[lane * 64 + c];
      ldsHln[lane * 64 + c] = (x - mu) * rstd * ln_g[c] + ln_b[c];
    }
  }
  __syncthreads();

  // k/v GEMMs: (16x64) @ Wk.T / Wv.T (64x64); k -> ldsH, v -> ldsV
  v16h A30 = frag_a_f32(ldsHln, 64, 0, lane);
  v16h A31 = frag_a_f32(ldsHln, 64, 32, lane);
#pragma unroll
  for (int j = 0; j < 4; ++j) {
    v8f ck = {};
    v16h Bk0 = frag_b_w(Wkh, 64, j * 16, 0, lane);
    ck = __builtin_amdgcn_wmma_f32_16x16x32_f16(false, A30, false, Bk0, (short)0, ck, false, false);
    v16h Bk1 = frag_b_w(Wkh, 64, j * 16, 32, lane);
    ck = __builtin_amdgcn_wmma_f32_16x16x32_f16(false, A31, false, Bk1, (short)0, ck, false, false);
    v8f cv = {};
    v16h Bv0 = frag_b_w(Wvh, 64, j * 16, 0, lane);
    cv = __builtin_amdgcn_wmma_f32_16x16x32_f16(false, A30, false, Bv0, (short)0, cv, false, false);
    v16h Bv1 = frag_b_w(Wvh, 64, j * 16, 32, lane);
    cv = __builtin_amdgcn_wmma_f32_16x16x32_f16(false, A31, false, Bv1, (short)0, cv, false, false);
#pragma unroll
    for (int r = 0; r < 8; ++r) {
      int m = r + (hi << 3);
      int col = j * 16 + n16;
      ldsH[m * 64 + col] = ck[r];
      ldsV[m * 64 + col] = cv[r];
    }
  }
  __syncthreads();

  // row norms: 1/max(||k||,1e-12); ||v|| to global
  if (lane < 16) {
    float k2 = 0.f, v2 = 0.f;
#pragma unroll 8
    for (int c = 0; c < 64; ++c) {
      float kk = ldsH[lane * 64 + c]; k2 += kk * kk;
      float vv = ldsV[lane * 64 + c]; v2 += vv * vv;
    }
    ldsS[lane] = 1.f / fmaxf(sqrtf(k2), 1e-12f);
    int pos = p0 + lane;
    if (pos < T_) vnorm_ws[(size_t)b * T_ + pos] = sqrtf(v2);
  }
  __syncthreads();

  // write kn, v (pos < L-1) and hlast (pos == L-1)
  for (int idx = lane; idx < 16 * 64; idx += 32) {
    int r = idx >> 6, c = idx & 63;
    int pos = p0 + r;
    if (pos < T_) {
      size_t o = ((size_t)b * T_ + pos) * 64 + c;
      kn_ws[o] = ldsH[idx] * ldsS[r];
      v_ws[o]  = ldsV[idx];
    } else {
      hlast[b * 64 + c] = ldsHln[idx];
    }
  }
}

// ---------------- Kernel B: sequential delta-rule scan --------------------
// One wave per batch; M (64x64) in VGPRs, 2 rows/lane; kn broadcast via LDS.
// Software-pipelined: while computing step t, the global loads for step t+1
// are in flight (their wait lands at the next iteration's ds_store, off the
// 1023-step dependency chain); prefetch walks t+2 into the caches.
__global__ __launch_bounds__(32) void scan_kernel(
    const float* __restrict__ kn_ws, const float* __restrict__ v_ws,
    const float* __restrict__ vnorm_ws, const float* __restrict__ log_thresh,
    float* __restrict__ Mws) {
  __shared__ float ldsKn[2][64];
  const int lane = threadIdx.x;
  const int b = blockIdx.x;
  const float thresh = __expf(log_thresh[0]);

  float m0[64], m1[64];
#pragma unroll
  for (int j = 0; j < 64; ++j) { m0[j] = 0.f; m1[j] = 0.f; }

  const float* knb = kn_ws + (size_t)b * T_ * 64;
  const float* vb  = v_ws  + (size_t)b * T_ * 64;
  const float* vnb = vnorm_ws + (size_t)b * T_;

  // prologue: loads for t = 0
  float ka = knb[lane], kc = knb[lane + 32];
  float va = vb[lane],  vc = vb[lane + 32];
  float vn = vnb[0];
  int cur = 0;

  for (int t = 0; t < T_; ++t) {
    ldsKn[cur][lane]      = ka;
    ldsKn[cur][lane + 32] = kc;
    float v0 = va, v1 = vc, vnt = vn;
    if (t + 1 < T_) {                       // issue t+1 loads; no wait yet
      const float* knp = knb + (size_t)(t + 1) * 64;
      const float* vp  = vb  + (size_t)(t + 1) * 64;
      ka = knp[lane]; kc = knp[lane + 32];
      va = vp[lane];  vc = vp[lane + 32];
      vn = vnb[t + 1];
      __builtin_prefetch(knp + 64, 0, 1);   // t+2 -> caches
      __builtin_prefetch(vp + 64, 0, 1);
    }
    __syncthreads();                         // S_NOP + ds wait (single wave)

    const float* kl = ldsKn[cur];
    float acc0 = 0.f, acc1 = 0.f;
#pragma unroll
    for (int j = 0; j < 64; ++j) { float kj = kl[j]; acc0 += m0[j] * kj; acc1 += m1[j] * kj; }
    float d0 = v0 - acc0, d1 = v1 - acc1;
    float ep = d0 * d0 + d1 * d1;
#pragma unroll
    for (int off = 16; off > 0; off >>= 1) ep += __shfl_xor(ep, off, 32);
    float err = sqrtf(ep);
    float g = 1.f / (1.f + __expf(-10.f * (err - thresh * vnt)));
    float s0 = g * d0, s1 = g * d1;
#pragma unroll
    for (int j = 0; j < 64; ++j) { float kj = kl[j]; m0[j] += s0 * kj; m1[j] += s1 * kj; }
    cur ^= 1;                                // double buffer: no trailing wait
  }

  float* Mb = Mws + (size_t)b * H_ * H_;
#pragma unroll
  for (int j = 0; j < 64; ++j) {
    Mb[lane * 64 + j]        = m0[j];
    Mb[(lane + 32) * 64 + j] = m1[j];
  }
}

// ---------------- Kernel C: head (q, Mq, Wr, Wo) --------------------------
__global__ __launch_bounds__(64) void head_kernel(
    const float* __restrict__ hlast, const float* __restrict__ Mws,
    const float* __restrict__ Wq, const float* __restrict__ Wr,
    const float* __restrict__ br, const float* __restrict__ Wo,
    const float* __restrict__ bo, float* __restrict__ out) {
  __shared__ float sh[64], sq[64], smq[64], sr[64];
  const int i = threadIdx.x, b = blockIdx.x;
  sh[i] = hlast[b * 64 + i];
  __syncthreads();
  float acc = 0.f;
  for (int j = 0; j < 64; ++j) acc += sh[j] * Wq[i * 64 + j];
  sq[i] = acc;
  __syncthreads();
  const float* Mb = Mws + (size_t)b * 4096 + i * 64;
  acc = 0.f;
  for (int j = 0; j < 64; ++j) acc += Mb[j] * sq[j];
  smq[i] = acc;
  __syncthreads();
  acc = br[i];
  for (int j = 0; j < 64; ++j) acc += smq[j] * Wr[i * 64 + j];
  sr[i] = acc;
  __syncthreads();
  acc = bo[i];
  for (int j = 0; j < 64; ++j) acc += sr[j] * Wo[i * 64 + j];
  out[b * 64 + i] = acc;
}

// ---------------- Launch ---------------------------------------------------
extern "C" void kernel_launch(void* const* d_in, const int* in_sizes, int n_in,
                              void* d_out, int out_size, void* d_ws, size_t ws_size,
                              hipStream_t stream) {
  (void)in_sizes; (void)n_in; (void)out_size; (void)ws_size;
  const int*   seq   = (const int*)  d_in[0];
  const float* logth = (const float*)d_in[1];
  const float* embed = (const float*)d_in[2];
  const float* W1    = (const float*)d_in[3];
  const float* b1    = (const float*)d_in[4];
  const float* W2    = (const float*)d_in[5];
  const float* b2    = (const float*)d_in[6];
  const float* lng   = (const float*)d_in[7];
  const float* lnb   = (const float*)d_in[8];
  const float* Wk    = (const float*)d_in[9];
  const float* Wv    = (const float*)d_in[10];
  const float* Wq    = (const float*)d_in[11];
  const float* Wr    = (const float*)d_in[12];
  const float* br    = (const float*)d_in[13];
  const float* Wo    = (const float*)d_in[14];
  const float* bo    = (const float*)d_in[15];

  char* ws = (char*)d_ws;
  _Float16* W1h  = (_Float16*)(ws + 0);          // 16384 B
  _Float16* W2h  = (_Float16*)(ws + 16384);      // 16384 B
  _Float16* Wkh  = (_Float16*)(ws + 32768);      //  8192 B
  _Float16* Wvh  = (_Float16*)(ws + 40960);      //  8192 B
  float* vnorm   = (float*)(ws + 49152);         // 128*1023*4 = 523776 B
  float* hlast   = (float*)(ws + 573440);        // 32768 B
  float* Mws     = (float*)(ws + 606208);        // 2097152 B
  float* kn_ws   = (float*)(ws + 2703360);       // 33521664 B
  float* v_ws    = (float*)(ws + 36225024);      // 33521664 B (end ~69.7 MB)

  hipLaunchKernelGGL(wconv_kernel, dim3(96), dim3(256), 0, stream,
                     W1, W2, Wk, Wv, W1h, W2h, Wkh, Wvh);
  hipLaunchKernelGGL(fused_token_kernel, dim3(B_ * (L_ / 16)), dim3(32), 0, stream,
                     seq, embed, b1, b2, lng, lnb, W1h, W2h, Wkh, Wvh,
                     kn_ws, v_ws, vnorm, hlast);
  hipLaunchKernelGGL(scan_kernel, dim3(B_), dim3(32), 0, stream,
                     kn_ws, v_ws, vnorm, logth, Mws);
  hipLaunchKernelGGL(head_kernel, dim3(B_), dim3(64), 0, stream,
                     hlast, Mws, Wq, Wr, br, Wo, bo, (float*)d_out);
}
